// memory_35167192219739
// MI455X (gfx1250) — matrix operand order repaired and checked by reading, not compiled, not run on hardware
//
#include <hip/hip_runtime.h>

#define BATCH   512
#define MEMSZ   131072
#define DIM     256
#define TOPK    256
#define NB1     512     // level-1 bins: float bits [30:22] (8 exp + 1 mant)
#define NB2     4096    // level-2 bins: float bits [21:10]
#define ROWS    64      // batch rows per workgroup (4 row-tiles of 16)
#define COLSEG  32      // grid.y segments over MEMSZ
#define COLS_PER_SEG (MEMSZ / COLSEG)   // 4096
#define ITERS   (COLS_PER_SEG / 64)     // 64 iterations of 64 cols

__constant__ const float kBETA  = 1e-8f;
__constant__ const float kALPHA = 0.5f;

typedef _Float16 v4h  __attribute__((ext_vector_type(4)));
typedef _Float16 v8h  __attribute__((ext_vector_type(8)));
typedef _Float16 v16h __attribute__((ext_vector_type(16)));
typedef float    v8f  __attribute__((ext_vector_type(8)));

__device__ __forceinline__ v16h cat8(v8h lo, v8h hi) {
  return __builtin_shufflevector(lo, hi, 0,1,2,3,4,5,6,7,8,9,10,11,12,13,14,15);
}

// Fused WMMA GEMM + score epilogue. MODE: 0=level-1 histogram,
// 1=accumulate-above + level-2 histogram, 2=boundary accumulate.
template <int MODE>
__global__ __launch_bounds__(256)
void score_pass(const float* __restrict__ q, const float* __restrict__ mk,
                const float* __restrict__ mhist, const float* __restrict__ mvals,
                unsigned* __restrict__ hist1, unsigned* __restrict__ hist2,
                const unsigned* __restrict__ thr1, const unsigned* __restrict__ rem2v,
                const unsigned* __restrict__ thr2, unsigned* __restrict__ cnt,
                float* __restrict__ sumJ, float* __restrict__ sumJV) {
  __shared__ _Float16 qs[ROWS][264];  // 64 q rows x 256 f16, padded (16B-aligned rows)
  __shared__ _Float16 ks[64][264];    // 64 key rows x 256 f16
  __shared__ unsigned lh[(MODE == 0) ? ROWS * NB1 : 16];
  __shared__ float    rs[ROWS][2];

  const int t     = threadIdx.x;
  const int wave  = t >> 5;
  const int lane  = t & 31;
  const int rwave = wave >> 1;      // row-tile 0..3 (16 rows each)
  const int cwave = wave & 1;       // col-pair 0..1 (32 cols each)
  const int mloc  = lane & 15;      // M (for A) / N (for B) within tile
  const int half  = lane >> 4;      // lane half selects K sub-group
  const int rowBase = blockIdx.x * ROWS;
  const int colSegBase = blockIdx.y * COLS_PER_SEG;

  if constexpr (MODE == 0) {
    for (int i = t; i < ROWS * NB1; i += 256) lh[i] = 0u;
  } else {
    if (t < 2 * ROWS) rs[t >> 1][t & 1] = 0.0f;
  }
  // stage q tile (f32 -> f16)
  for (int i = t; i < ROWS * DIM; i += 256) {
    const int r = i >> 8, c = i & 255;
    qs[r][c] = (_Float16)q[(rowBase + r) * DIM + c];
  }
  __syncthreads();

  // Hoist the wave-invariant A fragments into registers (8 x v16h = 64 VGPRs).
  // A 16x32 f16 layout: lane half h holds contiguous K = {h*16..h*16+15}.
  v16h a_frag[8];
#pragma unroll
  for (int kk = 0; kk < 8; ++kk) {
    const int kb = kk * 32;
    v8h a0 = *(const v8h*)(&qs[rwave * 16 + mloc][kb + half * 16]);
    v8h a1 = *(const v8h*)(&qs[rwave * 16 + mloc][kb + half * 16 + 8]);
    a_frag[kk] = cat8(a0, a1);
  }

  // B fragment loader: B 32x16 f16, lane half h holds contiguous K {h*16..h*16+15}
  const _Float16* brow0 = &ks[cwave * 32 + mloc][0];
  const _Float16* brow1 = &ks[cwave * 32 + 16 + mloc][0];

  for (int it = 0; it < ITERS; ++it) {
    const int colBase = colSegBase + it * 64;
    __syncthreads();  // previous-iteration LDS reads done
    // stage 64x256 key tile (f32 -> f16), float4 loads (L2-resident stream)
    for (int i = t; i < 64 * 64; i += 256) {
      const int r = i >> 6, c = (i & 63) * 4;
      const float4 f = *(const float4*)(mk + (size_t)(colBase + r) * DIM + c);
      v4h pk = { (_Float16)f.x, (_Float16)f.y, (_Float16)f.z, (_Float16)f.w };
      *(v4h*)&ks[r][c] = pk;
    }
    __syncthreads();

    // Two 16x16 tiles per wave, K=256: 16x v_wmma_f32_16x16x32_f16.
    // Software-pipelined B loads: prefetch kk+1 while kk WMMAs issue.
    v8f acc0 = {}, acc1 = {};
    v8h lo0 = *(const v8h*)(brow0 + half * 16);
    v8h hi0 = *(const v8h*)(brow0 + half * 16 + 8);
    v8h lo1 = *(const v8h*)(brow1 + half * 16);
    v8h hi1 = *(const v8h*)(brow1 + half * 16 + 8);
#pragma unroll
    for (int kk = 0; kk < 8; ++kk) {
      v8h nlo0, nhi0, nlo1, nhi1;
      if (kk < 7) {
        const int nkb = (kk + 1) * 32 + half * 16;
        nlo0 = *(const v8h*)(brow0 + nkb);
        nhi0 = *(const v8h*)(brow0 + nkb + 8);
        nlo1 = *(const v8h*)(brow1 + nkb);
        nhi1 = *(const v8h*)(brow1 + nkb + 8);
      }
      acc0 = __builtin_amdgcn_wmma_f32_16x16x32_f16(
          false, a_frag[kk], false, cat8(lo0, hi0), (short)0, acc0, false, false);
      acc1 = __builtin_amdgcn_wmma_f32_16x16x32_f16(
          false, a_frag[kk], false, cat8(lo1, hi1), (short)0, acc1, false, false);
      lo0 = nlo0; hi0 = nhi0; lo1 = nlo1; hi1 = nhi1;
    }

    // epilogue: C layout -> element r: M = r + 8*half, N = lane&15
#pragma unroll
    for (int tt = 0; tt < 2; ++tt) {
      const v8f& acc = tt ? acc1 : acc0;
      const int col = colBase + cwave * 32 + tt * 16 + mloc;
      const float hv = mhist[col];
      const float vv = (MODE == 0) ? 0.0f : mvals[col];
#pragma unroll
      for (int r8 = 0; r8 < 8; ++r8) {
        const int rloc = rwave * 16 + r8 + half * 8;
        const float e = __expf(acc[r8] - 1.0f);
        const float score = e * (hv + kBETA);
        const unsigned bits = __float_as_uint(score);
        const unsigned l1 = (bits >> 22) & (NB1 - 1);
        if constexpr (MODE == 0) {
          atomicAdd(&lh[rloc * NB1 + l1], 1u);
        } else {
          const int grow = rowBase + rloc;
          const unsigned t1 = thr1[grow];
          if constexpr (MODE == 1) {
            if (l1 > t1) {
              const float joint = e * (hv * kALPHA + kBETA);
              atomicAdd(&rs[rloc][0], joint);
              atomicAdd(&rs[rloc][1], joint * vv);
            } else if (l1 == t1) {
              const unsigned l2 = (bits >> 10) & (NB2 - 1);
              atomicAdd(&hist2[(size_t)grow * NB2 + l2], 1u);
            }
          } else {  // MODE 2
            if (l1 == t1) {
              const unsigned l2 = (bits >> 10) & (NB2 - 1);
              const unsigned t2 = thr2[grow];
              bool take = (l2 > t2);
              if (!take && l2 == t2) take = (atomicAdd(&cnt[grow], 1u) < rem2v[grow]);
              if (take) {
                const float joint = e * (hv * kALPHA + kBETA);
                atomicAdd(&rs[rloc][0], joint);
                atomicAdd(&rs[rloc][1], joint * vv);
              }
            }
          }
        }
      }
    }
  }
  __syncthreads();
  if constexpr (MODE == 0) {
    for (int i = t; i < ROWS * NB1; i += 256) {
      const unsigned c = lh[i];
      if (c) atomicAdd(&hist1[(size_t)(rowBase + (i >> 9)) * NB1 + (i & (NB1 - 1))], c);
    }
  } else {
    if (t < ROWS) {
      atomicAdd(&sumJ[rowBase + t],  rs[t][0]);
      atomicAdd(&sumJV[rowBase + t], rs[t][1]);
    }
  }
}

__global__ void zero_ws(unsigned* p, int n) {
  for (int i = blockIdx.x * blockDim.x + threadIdx.x; i < n; i += gridDim.x * blockDim.x)
    p[i] = 0u;
}

__global__ void select1(const unsigned* __restrict__ hist1,
                        unsigned* __restrict__ thr1, unsigned* __restrict__ rem1) {
  const int row = blockIdx.x * blockDim.x + threadIdx.x;
  if (row >= BATCH) return;
  const unsigned* h = hist1 + (size_t)row * NB1;
  unsigned cum = 0;
  for (int b = NB1 - 1; b >= 0; --b) {
    const unsigned c = h[b];
    if (cum + c >= (unsigned)TOPK) { thr1[row] = (unsigned)b; rem1[row] = TOPK - cum; return; }
    cum += c;
  }
  thr1[row] = 0u; rem1[row] = TOPK - cum;
}

__global__ void select2(const unsigned* __restrict__ hist2, const unsigned* __restrict__ rem1,
                        unsigned* __restrict__ thr2, unsigned* __restrict__ rem2) {
  const int row = blockIdx.x * blockDim.x + threadIdx.x;
  if (row >= BATCH) return;
  const unsigned* h = hist2 + (size_t)row * NB2;
  const unsigned R = rem1[row];
  unsigned cum = 0;
  for (int b = NB2 - 1; b >= 0; --b) {
    const unsigned c = h[b];
    if (cum + c >= R) { thr2[row] = (unsigned)b; rem2[row] = R - cum; return; }
    cum += c;
  }
  thr2[row] = 0u; rem2[row] = R - cum;
}

__global__ void finalize(const float* __restrict__ sumJ, const float* __restrict__ sumJV,
                         float* __restrict__ out) {
  const int row = blockIdx.x * blockDim.x + threadIdx.x;
  if (row < BATCH) out[row] = sumJV[row] / sumJ[row];
}

extern "C" void kernel_launch(void* const* d_in, const int* in_sizes, int n_in,
                              void* d_out, int out_size, void* d_ws, size_t ws_size,
                              hipStream_t stream) {
  const float* q     = (const float*)d_in[0];   // [512,256]
  const float* mk    = (const float*)d_in[1];   // [131072,256]
  const float* mvals = (const float*)d_in[2];   // [131072]
  const float* mhist = (const float*)d_in[3];   // [131072]
  float* out = (float*)d_out;

  // workspace layout (u32 units)
  unsigned* hist1 = (unsigned*)d_ws;            // 512*512
  unsigned* hist2 = hist1 + BATCH * NB1;        // 512*4096
  unsigned* thr1  = hist2 + (size_t)BATCH * NB2;
  unsigned* rem1  = thr1 + BATCH;
  unsigned* thr2  = rem1 + BATCH;
  unsigned* rem2  = thr2 + BATCH;
  unsigned* cnt   = rem2 + BATCH;
  float*    sumJ  = (float*)(cnt + BATCH);
  float*    sumJV = sumJ + BATCH;
  const int ws_u32 = BATCH * NB1 + BATCH * NB2 + 7 * BATCH;

  zero_ws<<<1024, 256, 0, stream>>>((unsigned*)d_ws, ws_u32);

  dim3 grid(BATCH / ROWS, COLSEG), block(256);
  score_pass<0><<<grid, block, 0, stream>>>(q, mk, mhist, mvals, hist1, hist2,
                                            thr1, rem2, thr2, cnt, sumJ, sumJV);
  select1<<<(BATCH + 255) / 256, 256, 0, stream>>>(hist1, thr1, rem1);
  score_pass<1><<<grid, block, 0, stream>>>(q, mk, mhist, mvals, hist1, hist2,
                                            thr1, rem2, thr2, cnt, sumJ, sumJV);
  select2<<<(BATCH + 255) / 256, 256, 0, stream>>>(hist2, rem1, thr2, rem2);
  score_pass<2><<<grid, block, 0, stream>>>(q, mk, mhist, mvals, hist1, hist2,
                                            thr1, rem2, thr2, cnt, sumJ, sumJV);
  finalize<<<(BATCH + 255) / 256, 256, 0, stream>>>(sumJ, sumJV, out);
}
